// _MelMFCC_43044162241268
// MI455X (gfx1250) — matrix operand order repaired and checked by reading, not compiled, not run on hardware
//
#include <hip/hip_runtime.h>
#include <hip/hip_bf16.h>

typedef __attribute__((ext_vector_type(16))) _Float16 v16h;
typedef __attribute__((ext_vector_type(8)))  _Float16 v8h;
typedef __attribute__((ext_vector_type(4)))  _Float16 v4h;
typedef __attribute__((ext_vector_type(8)))  float    v8f;

#define B_BATCH   64
#define WAVLEN    160000
#define TFRAMES   1001
#define MROWS     64064            // B_BATCH * TFRAMES
#define MPAD      64128            // 1002 * 64
#define KDFT      416              // 400 padded to 13*32
#define NDFT      416              // 402 padded
#define KMEL      224              // 208 padded to 7*32
#define NMEL      64
#define SPECP     224              // spec row pitch (halfs)
#define AP        40               // LDS row pitch in halfs (80B -> 16B aligned frags)

// ---- CDNA5 async global->LDS copy (ASYNCcnt-tracked), 16B per lane ----
__device__ __forceinline__ void async_copy_b128(const _Float16* gaddr, unsigned lds_off) {
  asm volatile("global_load_async_to_lds_b128 %0, %1, off"
               :: "v"(lds_off), "v"(gaddr)
               : "memory");
}
__device__ __forceinline__ void wait_async0() {
  asm volatile("s_wait_asynccnt 0" ::: "memory");
}
__device__ __forceinline__ unsigned lds_off32(const _Float16* p) {
  return (unsigned)(size_t)p;      // flat LDS aperture: addr[31:0] is the LDS byte address
}

// ---------------- constant builders (run on device every launch) ----------------

__global__ void k_basis(_Float16* __restrict__ basis) {
  int e = blockIdx.x * 256 + threadIdx.x;
  if (e >= KDFT * NDFT) return;
  int k = e / NDFT, n = e % NDFT;
  float v = 0.0f;
  if (k < 400 && n < 402) {
    int f = n >> 1;
    int kf = (k * f) % 400;                              // exact phase reduction
    float theta = 0.015707963267948967f * (float)kf;     // 2*pi/400
    float win = 0.5f - 0.5f * cosf(0.015707963267948967f * (float)k);
    float s = sinf(theta), c = cosf(theta);
    v = win * ((n & 1) ? s : c);                         // squared later: sign irrelevant
  }
  basis[e] = (_Float16)v;
}

__global__ void k_fb(_Float16* __restrict__ fb) {
  int e = blockIdx.x * 256 + threadIdx.x;
  if (e >= KMEL * NMEL) return;
  int k = e / NMEL, m = e % NMEL;
  float v = 0.0f;
  if (k < 201) {
    float freq = 40.0f * (float)k;
    float m_max = 2595.0f * log10f(1.0f + 8000.0f / 700.0f);
    float a0 = m_max * (float)(m)     * (1.0f / 65.0f) * (1.0f / 2595.0f);
    float a1 = m_max * (float)(m + 1) * (1.0f / 65.0f) * (1.0f / 2595.0f);
    float a2 = m_max * (float)(m + 2) * (1.0f / 65.0f) * (1.0f / 2595.0f);
    float f0 = 700.0f * (exp10f(a0) - 1.0f);
    float f1 = 700.0f * (exp10f(a1) - 1.0f);
    float f2 = 700.0f * (exp10f(a2) - 1.0f);
    float down = (freq - f0) / (f1 - f0);
    float up   = (f2 - freq) / (f2 - f1);
    v = fmaxf(0.0f, fminf(down, up));
  }
  fb[e] = (_Float16)v;
}

__global__ void k_frames(const float* __restrict__ wav, _Float16* __restrict__ frames) {
  long long e = (long long)blockIdx.x * 256 + threadIdx.x;
  if (e >= (long long)MPAD * KDFT) return;
  int m = (int)(e / KDFT);
  int k = (int)(e % KDFT);
  float v = 0.0f;
  if (m < MROWS && k < 400) {
    int b = m / TFRAMES;
    int t = m - b * TFRAMES;
    int j = 160 * t + k - 200;                           // reflect pad = 200
    if (j < 0) j = -j;
    if (j >= WAVLEN) j = 2 * (WAVLEN - 1) - j;
    v = wav[(size_t)b * WAVLEN + j];
  }
  frames[e] = (_Float16)v;
}

__global__ void k_zpad(_Float16* __restrict__ spec) {
  int e = blockIdx.x * 256 + threadIdx.x;                // zero spec cols 208..223
  if (e >= MPAD * 16) return;
  int m = e >> 4;
  spec[(size_t)m * SPECP + 208 + (e & 15)] = (_Float16)0.0f;
}

// ---------------- double-buffered WMMA GEMM core ----------------
// 8 waves: 4 (M) x 2 (N); workgroup tile 64x32; one 16x16x32 WMMA per wave per K-step.
// A tile: async global->LDS (ASYNCcnt). B tile: regs prefetch -> transposed scatter to next buffer.

template<int KT>
__device__ __forceinline__ v8f gemm_core(const _Float16* __restrict__ Ag, int apitch,
                                         const _Float16* __restrict__ Bg, int bpitch,
                                         int mBase, int nBase,
                                         _Float16* At0, _Float16* At1,
                                         _Float16* Bt0, _Float16* Bt1) {
  const int tid  = threadIdx.x;
  const int lane = tid & 31;
  const int wave = tid >> 5;
  const int wm = (wave & 3) * 16;
  const int wn = (wave >> 2) * 16;

  const int arow = tid >> 2, ac = (tid & 3) * 8;         // A: 64 rows x 32 halfs, 16B/thread
  const int bk = tid >> 3, bn4 = (tid & 7) * 4;          // B: 32x32 tile, 4 halfs/thread

  _Float16* Abuf[2] = { At0, At1 };
  _Float16* Bbuf[2] = { Bt0, Bt1 };

  const int r     = lane & 15;
  const int akoff = (lane >> 4) * 8;                     // A frag K {0..7,16..23} / {8..15,24..31}
  const int bkoff = (lane >> 4) * 8;                     // B frag K {0..7,16..23} / {8..15,24..31}

  // prologue: tile 0
  async_copy_b128(Ag + (size_t)(mBase + arow) * apitch + ac, lds_off32(At0 + arow * AP + ac));
  {
    v4h x = *(const v4h*)(Bg + (size_t)bk * bpitch + nBase + bn4);
    Bt0[(bn4 + 0) * AP + bk] = x[0];
    Bt0[(bn4 + 1) * AP + bk] = x[1];
    Bt0[(bn4 + 2) * AP + bk] = x[2];
    Bt0[(bn4 + 3) * AP + bk] = x[3];
  }
  wait_async0();
  __syncthreads();

  v8f acc = {};
  for (int kt = 0; kt < KT; ++kt) {
    const int cur = kt & 1, nxt = cur ^ 1;
    const bool pre = (kt + 1 < KT);
    v4h xb = {};
    if (pre) {        // overlap next tile's traffic with this tile's WMMA
      async_copy_b128(Ag + (size_t)(mBase + arow) * apitch + (kt + 1) * 32 + ac,
                      lds_off32(Abuf[nxt] + arow * AP + ac));
      xb = *(const v4h*)(Bg + (size_t)((kt + 1) * 32 + bk) * bpitch + nBase + bn4);
    }
    v8h alo = *(const v8h*)(Abuf[cur] + (wm + r) * AP + akoff);
    v8h ahi = *(const v8h*)(Abuf[cur] + (wm + r) * AP + akoff + 16);
    v8h blo = *(const v8h*)(Bbuf[cur] + (wn + r) * AP + bkoff);
    v8h bhi = *(const v8h*)(Bbuf[cur] + (wn + r) * AP + bkoff + 16);
    if (pre) {
      Bbuf[nxt][(bn4 + 0) * AP + bk] = xb[0];
      Bbuf[nxt][(bn4 + 1) * AP + bk] = xb[1];
      Bbuf[nxt][(bn4 + 2) * AP + bk] = xb[2];
      Bbuf[nxt][(bn4 + 3) * AP + bk] = xb[3];
    }
    v16h a, b;
#pragma unroll
    for (int j = 0; j < 8; ++j) { a[j] = alo[j]; a[8 + j] = ahi[j]; b[j] = blo[j]; b[8 + j] = bhi[j]; }
    acc = __builtin_amdgcn_wmma_f32_16x16x32_f16(false, a, false, b, (short)0, acc, false, false);
    wait_async0();
    __syncthreads();                                     // next buffer ready; safe to overwrite cur
  }
  return acc;
}

// GEMM 1: windowed DFT + |X|^2 epilogue
__global__ __launch_bounds__(256) void k_gemm_dft(const _Float16* __restrict__ frames,
                                                  const _Float16* __restrict__ basis,
                                                  _Float16* __restrict__ spec) {
  __shared__ __align__(16) _Float16 At[2][64 * AP];
  __shared__ __align__(16) _Float16 Bt[2][32 * AP];
  const int lane  = threadIdx.x & 31;
  const int wave  = threadIdx.x >> 5;
  const int mBase = blockIdx.x * 64;
  const int nBase = blockIdx.y * 32;

  v8f acc = gemm_core<13>(frames, KDFT, basis, NDFT, mBase, nBase,
                          At[0], At[1], Bt[0], Bt[1]);

  const int col     = nBase + ((wave >> 2) * 16) + (lane & 15);
  const int rowBase = mBase + ((wave & 3) * 16) + ((lane >> 4) * 8);
#pragma unroll
  for (int i = 0; i < 8; ++i) {
    float v = acc[i];
    float o = __shfl_xor(v, 1, 32);                      // re/im are adjacent lanes
    float p = v * v + o * o;
    if ((lane & 1) == 0)
      spec[(size_t)(rowBase + i) * SPECP + (col >> 1)] = (_Float16)p;
  }
}

// GEMM 2: mel projection
__global__ __launch_bounds__(256) void k_gemm_mel(const _Float16* __restrict__ spec,
                                                  const _Float16* __restrict__ fb,
                                                  float* __restrict__ mel) {
  __shared__ __align__(16) _Float16 At[2][64 * AP];
  __shared__ __align__(16) _Float16 Bt[2][32 * AP];
  const int lane  = threadIdx.x & 31;
  const int wave  = threadIdx.x >> 5;
  const int mBase = blockIdx.x * 64;
  const int nBase = blockIdx.y * 32;

  v8f acc = gemm_core<7>(spec, SPECP, fb, NMEL, mBase, nBase,
                         At[0], At[1], Bt[0], Bt[1]);

  const int col     = nBase + ((wave >> 2) * 16) + (lane & 15);
  const int rowBase = mBase + ((wave & 3) * 16) + ((lane >> 4) * 8);
#pragma unroll
  for (int i = 0; i < 8; ++i)
    mel[(size_t)(rowBase + i) * NMEL + col] = acc[i];
}

// ---------------- EMA: 4096 independent chains, lanes = adjacent mel channels ----------------

__global__ void k_ema(const float* __restrict__ mel, float* __restrict__ sm) {
  int g = blockIdx.x * blockDim.x + threadIdx.x;         // 0..4095
  int b = g >> 6, m = g & 63;
  const float* src = mel + ((size_t)b * TFRAMES) * NMEL + m;
  float*       dst = sm  + ((size_t)b * TFRAMES) * NMEL + m;
  float s = 0.0f;
  for (int t = 0; t < TFRAMES; ++t) {
    float x = src[(size_t)t * NMEL];
    s = fmaf(0.98f, s, 0.02f * x);
    dst[(size_t)t * NMEL] = s;
  }
}

// ---------------- final: mel_norm + MFCC (DCT built in LDS) ----------------

__global__ __launch_bounds__(256) void k_final(const float* __restrict__ mel,
                                               const float* __restrict__ sm,
                                               float* __restrict__ out) {
  __shared__ float dct[64 * 20];
  int tid = threadIdx.x;
  for (int i = tid; i < 64 * 20; i += 256) {
    int m = i / 20, k = i % 20;
    float v = 0.17677669529663687f * cosf(0.04908738521234052f * ((float)m + 0.5f) * (float)k);
    if (k == 0) v = 0.125f;                              // sqrt(1/64)
    dct[i] = v;
  }
  __syncthreads();
  int b = blockIdx.x;
  float* outMel  = out;                                  // [64][64][1001]
  float* outMfcc = out + (size_t)B_BATCH * 64 * TFRAMES; // [64][20][1001]
  for (int t = tid; t < TFRAMES; t += 256) {
    size_t row = ((size_t)b * TFRAMES + t) * NMEL;
    float db[64];
#pragma unroll
    for (int m = 0; m < 64; ++m) {
      float x = mel[row + m];
      float s = sm[row + m];
      outMel[((size_t)b * 64 + m) * TFRAMES + t] = log1pf(x / (s + 1e-6f));
      db[m] = 10.0f * log10f(fmaxf(x, 1e-10f));
    }
    for (int k = 0; k < 20; ++k) {
      float acc = 0.0f;
#pragma unroll
      for (int m = 0; m < 64; ++m) acc = fmaf(db[m], dct[m * 20 + k], acc);
      outMfcc[((size_t)b * 20 + k) * TFRAMES + t] = acc;
    }
  }
}

// ---------------- launcher ----------------

extern "C" void kernel_launch(void* const* d_in, const int* in_sizes, int n_in,
                              void* d_out, int out_size, void* d_ws, size_t ws_size,
                              hipStream_t stream) {
  const float* wav = (const float*)d_in[0];
  float* out = (float*)d_out;

  char* ws = (char*)d_ws;
  size_t off = 0;
  auto take = [&](size_t bytes) { void* p = ws + off; off += (bytes + 255) & ~(size_t)255; return p; };

  _Float16* frames = (_Float16*)take((size_t)MPAD * KDFT * 2);   // 53.4 MB
  _Float16* basis  = (_Float16*)take((size_t)KDFT * NDFT * 2);   // 0.35 MB
  _Float16* spec   = (_Float16*)take((size_t)MPAD * SPECP * 2);  // 28.7 MB
  _Float16* fb     = (_Float16*)take((size_t)KMEL * NMEL * 2);   // 28 KB
  float*    mel    = (float*)take((size_t)MPAD * NMEL * 4);      // 16.4 MB
  float*    sm     = (float*)take((size_t)MPAD * NMEL * 4);      // 16.4 MB

  k_basis<<<(KDFT * NDFT + 255) / 256, 256, 0, stream>>>(basis);
  k_fb<<<(KMEL * NMEL + 255) / 256, 256, 0, stream>>>(fb);
  k_frames<<<(int)(((long long)MPAD * KDFT + 255) / 256), 256, 0, stream>>>(wav, frames);
  k_zpad<<<(MPAD * 16 + 255) / 256, 256, 0, stream>>>(spec);
  k_gemm_dft<<<dim3(MPAD / 64, NDFT / 32), 256, 0, stream>>>(frames, basis, spec);
  k_gemm_mel<<<dim3(MPAD / 64, NMEL / 32), 256, 0, stream>>>(spec, fb, mel);
  k_ema<<<16, 256, 0, stream>>>(mel, sm);
  k_final<<<B_BATCH, 256, 0, stream>>>(mel, sm, out);
}
